// L2Embedding_47047071760843
// MI455X (gfx1250) — matrix operand order repaired and checked by reading, not compiled, not run on hardware
//
#include <hip/hip_runtime.h>
#include <hip/hip_bf16.h>

// CDNA5 / gfx1250 fused VQ codebook kernel.
// p_code = softmax_V( relu(temp) * (2*x.yT - |x|^2 - |y|^2) ), new_latent = table[argmax].
// fp32-accurate GEMM via f16 hi/lo split: dot = Ah*Bh + Ah*Bl + Al*Bh using
// v_wmma_f32_16x16x32_f16 (wave32 WMMA).

typedef _Float16 v16h __attribute__((ext_vector_type(16)));
typedef _Float16 v8h  __attribute__((ext_vector_type(8)));
typedef float    v8f  __attribute__((ext_vector_type(8)));

#define DIMD   256
#define DIMV   1024
#define NROWS  65536      // B*S
#define MTILE  16
#define NWAVES 8
#define NPW    128        // columns per wave
#define NT     8          // 16-wide N tiles per wave

// ---------------------------------------------------------------------------
// Prep: table[1024,256] f32 -> f16 hi/lo planes + exact fp32 row norms |y|^2.
// ---------------------------------------------------------------------------
__global__ __launch_bounds__(DIMD)
void vq_prep_table(const float* __restrict__ table,
                   _Float16* __restrict__ th,
                   _Float16* __restrict__ tl,
                   float* __restrict__ ysq)
{
    const int v = blockIdx.x;
    const int t = threadIdx.x;
    const float val = table[(size_t)v * DIMD + t];
    const _Float16 h  = (_Float16)val;
    const _Float16 lo = (_Float16)(val - (float)h);
    th[(size_t)v * DIMD + t] = h;
    tl[(size_t)v * DIMD + t] = lo;

    float sq = val * val;
    #pragma unroll
    for (int off = 16; off >= 1; off >>= 1) sq += __shfl_xor(sq, off);

    __shared__ float part[NWAVES];
    if ((t & 31) == 0) part[t >> 5] = sq;
    __syncthreads();
    if (t == 0) {
        float s = 0.f;
        #pragma unroll
        for (int i = 0; i < NWAVES; ++i) s += part[i];
        ysq[v] = s;
    }
}

// ---------------------------------------------------------------------------
// Fused GEMM + softmax + argmax + code gather.
// One block = 16 rows of x, all 1024 codebook columns. 8 waves, 128 cols each.
// ---------------------------------------------------------------------------
__global__ __launch_bounds__(256)
void vq_fused(const float* __restrict__ x,       // [NROWS, D]
              const float* __restrict__ table,   // [V, D] fp32 (for gather)
              const float* __restrict__ temp,    // [1]
              const _Float16* __restrict__ th,   // [V, D] hi plane
              const _Float16* __restrict__ tl,   // [V, D] lo plane
              const float* __restrict__ ysq,     // [V]
              float* __restrict__ out_p,         // [NROWS, V]
              float* __restrict__ out_lat)       // [NROWS, D]
{
    __shared__ __attribute__((aligned(16))) _Float16 sXh[MTILE * DIMD];
    __shared__ __attribute__((aligned(16))) _Float16 sXl[MTILE * DIMD];
    __shared__ float sXsq[MTILE];
    __shared__ float sRedVal[NWAVES][MTILE];
    __shared__ int   sRedIdx[NWAVES][MTILE];
    __shared__ float sRedSum[NWAVES][MTILE];
    __shared__ float sMax[MTILE];
    __shared__ int   sIdx[MTILE];
    __shared__ float sInv[MTILE];

    const int tid   = threadIdx.x;
    const int wave  = tid >> 5;
    const int lane  = tid & 31;
    const int l16   = lane & 15;
    const int lhalf = lane >> 4;          // 0: lanes 0-15, 1: lanes 16-31
    const int m0    = blockIdx.x * MTILE;

    // ---- Stage x tile into LDS as f16 hi/lo; fp32 row norms via shfl reduce.
    {
        const int row  = tid >> 4;          // 16 threads per row
        const int cseg = (tid & 15) << 4;   // 16-float chunk
        const float* xr = x + (size_t)(m0 + row) * DIMD + cseg;
        float sq = 0.f;
        #pragma unroll
        for (int j = 0; j < 16; ++j) {
            const float vv = xr[j];
            const _Float16 h  = (_Float16)vv;
            const _Float16 lo = (_Float16)(vv - (float)h);
            sXh[row * DIMD + cseg + j] = h;
            sXl[row * DIMD + cseg + j] = lo;
            sq += vv * vv;
        }
        #pragma unroll
        for (int off = 8; off >= 1; off >>= 1) sq += __shfl_xor(sq, off);
        if ((tid & 15) == 0) sXsq[row] = sq;
    }
    __syncthreads();

    // ---- GEMM: acc[t] += x . table^T over K=256, split-f16 (3 WMMAs / K-step)
    v8f acc[NT];
    #pragma unroll
    for (int t = 0; t < NT; ++t) acc[t] = {};

    const int koffA = lhalf * 8;    // A frag: K 0-7/16-23 vs 8-15/24-31
    const int koffB = lhalf * 8;    // B frag: K 0-7/16-23 vs 8-15/24-31
    const int nbase = wave * NPW + l16;

    for (int kk = 0; kk < 8; ++kk) {
        const int k0 = kk * 32;
        union { v16h v; v8h h[2]; } ah, al;
        ah.h[0] = *(const v8h*)&sXh[l16 * DIMD + k0 + koffA];
        ah.h[1] = *(const v8h*)&sXh[l16 * DIMD + k0 + koffA + 16];
        al.h[0] = *(const v8h*)&sXl[l16 * DIMD + k0 + koffA];
        al.h[1] = *(const v8h*)&sXl[l16 * DIMD + k0 + koffA + 16];
        #pragma unroll
        for (int t = 0; t < NT; ++t) {
            const int n = nbase + t * 16;
            union { v16h v; v8h h[2]; } bh, bl;
            const _Float16* rh = th + (size_t)n * DIMD + k0 + koffB;
            const _Float16* rl = tl + (size_t)n * DIMD + k0 + koffB;
            bh.h[0] = *(const v8h*)(rh);
            bh.h[1] = *(const v8h*)(rh + 16);
            bl.h[0] = *(const v8h*)(rl);
            bl.h[1] = *(const v8h*)(rl + 16);
            acc[t] = __builtin_amdgcn_wmma_f32_16x16x32_f16(
                false, ah.v, false, bh.v, (short)0, acc[t], false, false);
            acc[t] = __builtin_amdgcn_wmma_f32_16x16x32_f16(
                false, ah.v, false, bl.v, (short)0, acc[t], false, false);
            acc[t] = __builtin_amdgcn_wmma_f32_16x16x32_f16(
                false, al.v, false, bh.v, (short)0, acc[t], false, false);
        }
    }

    // ---- similarity + per-row running max/argmax (C layout: M=r+8*lhalf, N=l16)
    const float s0 = fmaxf(temp[0], 0.f);   // relu(temp)
    float xsqv[8];
    #pragma unroll
    for (int r = 0; r < 8; ++r) xsqv[r] = sXsq[r + 8 * lhalf];

    float mval[8]; int midx[8];
    #pragma unroll
    for (int r = 0; r < 8; ++r) { mval[r] = -3.4e38f; midx[r] = 0; }

    #pragma unroll
    for (int t = 0; t < NT; ++t) {
        const int n = nbase + t * 16;
        const float yq = ysq[n];
        #pragma unroll
        for (int r = 0; r < 8; ++r) {
            const float sim = s0 * (2.f * acc[t][r] - xsqv[r] - yq);
            acc[t][r] = sim;
            if (sim > mval[r]) { mval[r] = sim; midx[r] = n; }
        }
    }
    // reduce (max, idx) over the 16 lanes sharing each row (wave32-safe)
    #pragma unroll
    for (int off = 8; off >= 1; off >>= 1) {
        #pragma unroll
        for (int r = 0; r < 8; ++r) {
            const float ov = __shfl_xor(mval[r], off);
            const int   oi = __shfl_xor(midx[r], off);
            if (ov > mval[r] || (ov == mval[r] && oi < midx[r])) {
                mval[r] = ov; midx[r] = oi;
            }
        }
    }
    if (l16 == 0) {
        #pragma unroll
        for (int r = 0; r < 8; ++r) {
            sRedVal[wave][r + 8 * lhalf] = mval[r];
            sRedIdx[wave][r + 8 * lhalf] = midx[r];
        }
    }
    __syncthreads();
    if (tid < MTILE) {
        float bv = sRedVal[0][tid]; int bi = sRedIdx[0][tid];
        #pragma unroll
        for (int wv = 1; wv < NWAVES; ++wv) {
            const float vv = sRedVal[wv][tid]; const int ii = sRedIdx[wv][tid];
            if (vv > bv || (vv == bv && ii < bi)) { bv = vv; bi = ii; }
        }
        sMax[tid] = bv; sIdx[tid] = bi;
    }
    __syncthreads();

    // ---- exp + row sum
    float rowmax[8];
    #pragma unroll
    for (int r = 0; r < 8; ++r) rowmax[r] = sMax[r + 8 * lhalf];
    float rs[8];
    #pragma unroll
    for (int r = 0; r < 8; ++r) rs[r] = 0.f;
    #pragma unroll
    for (int t = 0; t < NT; ++t) {
        #pragma unroll
        for (int r = 0; r < 8; ++r) {
            const float e = __expf(acc[t][r] - rowmax[r]);
            acc[t][r] = e;
            rs[r] += e;
        }
    }
    #pragma unroll
    for (int off = 8; off >= 1; off >>= 1) {
        #pragma unroll
        for (int r = 0; r < 8; ++r) rs[r] += __shfl_xor(rs[r], off);
    }
    if (l16 == 0) {
        #pragma unroll
        for (int r = 0; r < 8; ++r) sRedSum[wave][r + 8 * lhalf] = rs[r];
    }
    __syncthreads();
    if (tid < MTILE) {
        float s = 0.f;
        #pragma unroll
        for (int wv = 0; wv < NWAVES; ++wv) s += sRedSum[wv][tid];
        sInv[tid] = 1.f / s;
    }
    __syncthreads();

    // ---- normalize + store p_code
    float rinv[8];
    #pragma unroll
    for (int r = 0; r < 8; ++r) rinv[r] = sInv[r + 8 * lhalf];
    #pragma unroll
    for (int t = 0; t < NT; ++t) {
        const int n = nbase + t * 16;
        #pragma unroll
        for (int r = 0; r < 8; ++r) {
            out_p[(size_t)(m0 + r + 8 * lhalf) * DIMV + n] = acc[t][r] * rinv[r];
        }
    }

    // ---- new_latent forward value = table[argmax] (straight-through)
    {
        const int row = tid >> 4;
        const int idx = sIdx[row];
        const float* src = table + (size_t)idx * DIMD;
        float* dst = out_lat + (size_t)(m0 + row) * DIMD;
        #pragma unroll
        for (int j = 0; j < 16; ++j) {
            const int c = (tid & 15) + j * 16;
            dst[c] = src[c];
        }
    }
}

// ---------------------------------------------------------------------------
extern "C" void kernel_launch(void* const* d_in, const int* in_sizes, int n_in,
                              void* d_out, int out_size, void* d_ws, size_t ws_size,
                              hipStream_t stream)
{
    const float* x     = (const float*)d_in[0];   // enc_embs [32,2048,256]
    const float* table = (const float*)d_in[1];   // [1024,256]
    const float* temp  = (const float*)d_in[2];   // [1]
    // d_in[3] = first_n_real_mel == 0 -> reference path handled here.

    _Float16* th  = (_Float16*)d_ws;                     // 512 KB
    _Float16* tl  = th + (size_t)DIMV * DIMD;            // 512 KB
    float*    ysq = (float*)(tl + (size_t)DIMV * DIMD);  // 4 KB

    float* out_p   = (float*)d_out;                      // [65536,1024]
    float* out_lat = out_p + (size_t)NROWS * DIMV;       // [65536,256]

    vq_prep_table<<<DIMV, DIMD, 0, stream>>>(table, th, tl, ysq);
    vq_fused<<<NROWS / MTILE, 256, 0, stream>>>(x, table, temp, th, tl, ysq,
                                                out_p, out_lat);
}
